// MLPLayer_37744172597623
// MI455X (gfx1250) — compile-verified, matrix operand order checked by reading, not run on hardware
//
#include <hip/hip_runtime.h>
#include <hip/hip_bf16.h>

// ---------------------------------------------------------------------------
// Problem constants (from the reference)
// ---------------------------------------------------------------------------
#define BATCH 8
#define SEQ   4096
#define CIN   1024            // C
#define HID   4096            // H
#define EMB   256             // E
#define MROWS (BATCH * SEQ)   // 32768 token rows

// GEMM tiling: 128x128 block tile, 4 waves (2x2), each wave 64x64 (4x4 WMMA)
#define BLK      128
#define TILE_K   32
#define LDSS     40           // padded LDS row stride (bf16 elements)
#define BUFSZ    (BLK * LDSS) // one LDS buffer (elements)

typedef __attribute__((ext_vector_type(16))) __bf16 bf16x16;
typedef __attribute__((ext_vector_type(8)))  float  floatx8;
typedef int nat_int4 __attribute__((vector_size(16)));   // native vector, matches builtin param

union FragAB {
    uint4   q[2];   // 32 bytes
    bf16x16 v;      // 16 bf16 = 8 VGPRs
};

// Async global->LDS (CDNA5): builtin-guarded; falls back to sync staging.
#ifndef __has_builtin
#define __has_builtin(x) 0
#endif
#if defined(__HIP_DEVICE_COMPILE__) && __has_builtin(__builtin_amdgcn_global_load_async_to_lds_b128)
#define USE_ASYNC 1
#else
#define USE_ASYNC 0
#endif

// builtin signature (from hipcc diagnostics):
//   (AS1 nat_int4* src, AS3 nat_int4* dst, Ii offset, Ii cpol)
#define GPTR(p) ((__attribute__((address_space(1))) nat_int4*)(const void*)(p))
#define LPTR(p) ((__attribute__((address_space(3))) nat_int4*)(p))

__device__ __forceinline__ void wait_async0() {
#if USE_ASYNC
#if __has_builtin(__builtin_amdgcn_s_wait_asynccnt)
    __builtin_amdgcn_s_wait_asynccnt(0);
#else
    asm volatile("s_wait_asynccnt 0x0" ::: "memory");
#endif
#endif
}

__device__ __forceinline__ unsigned short f32_to_bf16_rne(float f) {
    unsigned int u = __float_as_uint(f);
    unsigned int r = u + 0x7FFFu + ((u >> 16) & 1u);
    return (unsigned short)(r >> 16);
}

// ---------------------------------------------------------------------------
// Kernel 1: mod = embedding @ W_emb + b_emb   -> scale/shift [B, 2C]  (tiny)
// ---------------------------------------------------------------------------
__global__ __launch_bounds__(256)
void mod_kernel(const float* __restrict__ emb,    // [B, E]
                const float* __restrict__ W_emb,  // [E, 2C]
                const float* __restrict__ b_emb,  // [2C]
                float* __restrict__ ss)           // [B, 2C]
{
    int c = blockIdx.x * 256 + threadIdx.x;   // 0 .. 2C-1
    int b = blockIdx.y;
    const float* e = emb + b * EMB;
    float acc = b_emb[c];
    #pragma unroll 8
    for (int k = 0; k < EMB; ++k)
        acc += e[k] * W_emb[(size_t)k * (2 * CIN) + c];
    ss[(size_t)b * (2 * CIN) + c] = acc;
}

// ---------------------------------------------------------------------------
// Kernel 2: branch = LN(x) * (scale+1) + shift, emitted as bf16.
// ---------------------------------------------------------------------------
__global__ __launch_bounds__(256)
void ln_mod_kernel(const float* __restrict__ x,   // [M, C]
                   const float* __restrict__ ss,  // [B, 2C]
                   unsigned short* __restrict__ outb) // [M, C] bf16
{
    int row = blockIdx.x;
    int b   = row >> 12;                 // 4096 rows per batch
    int tid = threadIdx.x;

    const float4* xr = (const float4*)(x + (size_t)row * CIN);
    float4 v = xr[tid];
    float s  = v.x + v.y + v.z + v.w;
    float s2 = v.x * v.x + v.y * v.y + v.z * v.z + v.w * v.w;

    __shared__ float r1[256];
    __shared__ float r2[256];
    r1[tid] = s; r2[tid] = s2;
    __syncthreads();
    #pragma unroll
    for (int st = 128; st > 0; st >>= 1) {
        if (tid < st) { r1[tid] += r1[tid + st]; r2[tid] += r2[tid + st]; }
        __syncthreads();
    }
    float mu  = r1[0] * (1.0f / CIN);
    float var = r2[0] * (1.0f / CIN) - mu * mu;
    float inv = rsqrtf(var + 1e-5f);

    int c0 = tid * 4;
    const float* sc = ss + (size_t)b * (2 * CIN);
    float o0 = (v.x - mu) * inv * (sc[c0 + 0] + 1.0f) + sc[CIN + c0 + 0];
    float o1 = (v.y - mu) * inv * (sc[c0 + 1] + 1.0f) + sc[CIN + c0 + 1];
    float o2 = (v.z - mu) * inv * (sc[c0 + 2] + 1.0f) + sc[CIN + c0 + 2];
    float o3 = (v.w - mu) * inv * (sc[c0 + 3] + 1.0f) + sc[CIN + c0 + 3];

    union { unsigned short h[4]; uint2 u; } pk;
    pk.h[0] = f32_to_bf16_rne(o0);
    pk.h[1] = f32_to_bf16_rne(o1);
    pk.h[2] = f32_to_bf16_rne(o2);
    pk.h[3] = f32_to_bf16_rne(o3);
    *(uint2*)(outb + (size_t)row * CIN + c0) = pk.u;
}

// ---------------------------------------------------------------------------
// Kernel 3: transpose + fp32->bf16 convert: dst[c][r] = bf16(src[r][c])
// ---------------------------------------------------------------------------
__global__ __launch_bounds__(256)
void transpose_bf16(const float* __restrict__ src, int R, int Cn,
                    unsigned short* __restrict__ dst)
{
    __shared__ float tile[32][33];
    int tx = threadIdx.x, ty = threadIdx.y;
    int bx = blockIdx.x * 32, by = blockIdx.y * 32;

    #pragma unroll
    for (int l = 0; l < 4; ++l)
        tile[ty + l * 8][tx] = src[(size_t)(by + ty + l * 8) * Cn + bx + tx];
    __syncthreads();
    #pragma unroll
    for (int l = 0; l < 4; ++l)
        dst[(size_t)(bx + ty + l * 8) * R + by + tx] =
            f32_to_bf16_rne(tile[tx][ty + l * 8]);
}

// ---------------------------------------------------------------------------
// Kernel 4: bf16 WMMA GEMM  D = A[M,K] * Bt[N,K]^T  (+ fused epilogue)
//   MODE 0: val = gelu_exact(acc + b1[n])          -> bf16 h
//   MODE 1: out = resid + gamma[n] * (acc + b2[n]) -> fp32 out
// 4 waves/block (2x2), 64x64 per wave; double-buffered LDS; async staging.
// ---------------------------------------------------------------------------
template <int MODE>
__global__ __launch_bounds__(128)
void gemm_bf16_wmma(const unsigned short* __restrict__ A,   // [M,K] bf16
                    const unsigned short* __restrict__ Bt,  // [N,K] bf16
                    int M, int N, int K,
                    const float* __restrict__ bias,         // [N]
                    const float* __restrict__ resid,        // [M,N] (MODE 1)
                    const float* __restrict__ gamma,        // [N]   (MODE 1)
                    unsigned short* __restrict__ outB,      // [M,N] (MODE 0)
                    float* __restrict__ outF)               // [M,N] (MODE 1)
{
    __shared__ unsigned short sA[2 * BUFSZ];
    __shared__ unsigned short sB[2 * BUFSZ];

    const int tid   = threadIdx.x;       // 0..127
    const int lane  = tid & 31;
    const int wave  = tid >> 5;          // 0..3
    const int waveM = wave >> 1;         // 0..1 -> 64-row slab
    const int waveN = wave & 1;          // 0..1 -> 64-col slab
    const int blockN = blockIdx.x * BLK;
    const int blockM = blockIdx.y * BLK;

    floatx8 acc[4][4];
    #pragma unroll
    for (int i = 0; i < 4; ++i)
        #pragma unroll
        for (int j = 0; j < 4; ++j)
            acc[i][j] = (floatx8){0.f, 0.f, 0.f, 0.f, 0.f, 0.f, 0.f, 0.f};

    // cooperative staging: one full 32-elem K-row of A and of B per thread
    const unsigned short* aRow = A  + (size_t)(blockM + tid) * K;
    const unsigned short* bRow = Bt + (size_t)(blockN + tid) * K;
    unsigned short* laRow = sA + tid * LDSS;
    unsigned short* lbRow = sB + tid * LDSS;

    // fragment addressing per ISA 16-bit A/B layout:
    // lanes 0-15 hold K 0-7 (v0..3) & 16-23 (v4..7); lanes 16-31 hold 8-15 & 24-31
    const int kb   = (lane < 16) ? 0 : 8;
    const int rsel = lane & 15;
    const int iters = K / TILE_K;

    auto stage = [&](int k0, int buf) {
        const unsigned short* ga = aRow + k0;
        const unsigned short* gb = bRow + k0;
        unsigned short* la = laRow + buf * BUFSZ;
        unsigned short* lb = lbRow + buf * BUFSZ;
#if USE_ASYNC
        #pragma unroll
        for (int c = 0; c < 4; ++c) {
            __builtin_amdgcn_global_load_async_to_lds_b128(
                GPTR(ga + c * 8), LPTR(la + c * 8), 0, 0);
            __builtin_amdgcn_global_load_async_to_lds_b128(
                GPTR(gb + c * 8), LPTR(lb + c * 8), 0, 0);
        }
#else
        uint4 va[4], vb[4];
        #pragma unroll
        for (int c = 0; c < 4; ++c) {
            va[c] = ((const uint4*)ga)[c];
            vb[c] = ((const uint4*)gb)[c];
        }
        #pragma unroll
        for (int c = 0; c < 4; ++c) {
            ((uint4*)la)[c] = va[c];
            ((uint4*)lb)[c] = vb[c];
        }
#endif
    };

    stage(0, 0);

    for (int k = 0; k < iters; ++k) {
        const int cur = k & 1;
        wait_async0();          // tile k fully landed in LDS (this wave's part)
        __syncthreads();        // all waves' tile k visible; buf cur^1 free
        if (k + 1 < iters)
            stage((k + 1) * TILE_K, cur ^ 1);  // overlap HBM with WMMA below

        const unsigned short* baseA = sA + cur * BUFSZ;
        const unsigned short* baseB = sB + cur * BUFSZ;

        FragAB fa[4], fb[4];
        #pragma unroll
        for (int i = 0; i < 4; ++i) {
            const unsigned short* p = baseA + (waveM * 64 + i * 16 + rsel) * LDSS + kb;
            fa[i].q[0] = *(const uint4*)(p);
            fa[i].q[1] = *(const uint4*)(p + 16);
        }
        #pragma unroll
        for (int j = 0; j < 4; ++j) {
            const unsigned short* p = baseB + (waveN * 64 + j * 16 + rsel) * LDSS + kb;
            fb[j].q[0] = *(const uint4*)(p);
            fb[j].q[1] = *(const uint4*)(p + 16);
        }
        #pragma unroll
        for (int i = 0; i < 4; ++i)
            #pragma unroll
            for (int j = 0; j < 4; ++j)
                acc[i][j] = __builtin_amdgcn_wmma_f32_16x16x32_bf16(
                    false, fa[i].v, false, fb[j].v,
                    (short)0, acc[i][j], false, false);
    }

    // Epilogue — C/D layout: lane(0-15)=N, VGPR v -> M=v (lanes<16) / v+8
    const int mOut = blockM + waveM * 64;
    const int nOut = blockN + waveN * 64;
    const int col  = lane & 15;
    const int mo   = (lane >> 4) * 8;

    #pragma unroll
    for (int i = 0; i < 4; ++i) {
        #pragma unroll
        for (int j = 0; j < 4; ++j) {
            const int n  = nOut + j * 16 + col;
            const float bv = bias[n];
            #pragma unroll
            for (int v = 0; v < 8; ++v) {
                const int m = mOut + i * 16 + mo + v;
                float val = acc[i][j][v] + bv;
                if (MODE == 0) {
                    // exact (erf) GELU
                    val = 0.5f * val * (1.0f + erff(val * 0.70710678118654752f));
                    outB[(size_t)m * N + n] = f32_to_bf16_rne(val);
                } else {
                    const size_t idx = (size_t)m * N + n;
                    outF[idx] = resid[idx] + gamma[n] * val;
                }
            }
        }
    }
}

// ---------------------------------------------------------------------------
// Launcher
// ---------------------------------------------------------------------------
extern "C" void kernel_launch(void* const* d_in, const int* in_sizes, int n_in,
                              void* d_out, int out_size, void* d_ws, size_t ws_size,
                              hipStream_t stream)
{
    const float* in_tensor = (const float*)d_in[0];
    const float* embedding = (const float*)d_in[1];
    const float* W_emb     = (const float*)d_in[2];
    const float* b_emb     = (const float*)d_in[3];
    const float* W1        = (const float*)d_in[4];
    const float* b1        = (const float*)d_in[5];
    const float* W2        = (const float*)d_in[6];
    const float* b2        = (const float*)d_in[7];
    const float* gamma     = (const float*)d_in[8];
    float* out = (float*)d_out;

    // workspace carve-up
    char* ws = (char*)d_ws;
    size_t off = 0;
    float* ss = (float*)(ws + off);                    off += (size_t)BATCH * 2 * CIN * 4; // 64 KB
    unsigned short* Xb  = (unsigned short*)(ws + off); off += (size_t)MROWS * CIN * 2;     // 64 MB
    unsigned short* W1t = (unsigned short*)(ws + off); off += (size_t)HID * CIN * 2;       // 8 MB
    unsigned short* W2t = (unsigned short*)(ws + off); off += (size_t)CIN * HID * 2;       // 8 MB
    unsigned short* Hb  = (unsigned short*)(ws + off); off += (size_t)MROWS * HID * 2;     // 256 MB

    // 1. scale/shift
    mod_kernel<<<dim3((2 * CIN) / 256, BATCH), 256, 0, stream>>>(embedding, W_emb, b_emb, ss);

    // 2. LayerNorm + adaLN modulation -> bf16 activations
    ln_mod_kernel<<<MROWS, 256, 0, stream>>>(in_tensor, ss, Xb);

    // 3. weight convert+transpose: W1 [C,H] -> W1t [H,C]; W2 [H,C] -> W2t [C,H]
    transpose_bf16<<<dim3(HID / 32, CIN / 32), dim3(32, 8), 0, stream>>>(W1, CIN, HID, W1t);
    transpose_bf16<<<dim3(CIN / 32, HID / 32), dim3(32, 8), 0, stream>>>(W2, HID, CIN, W2t);

    // 4. GEMM1 + GELU -> h (bf16)
    gemm_bf16_wmma<0><<<dim3(HID / BLK, MROWS / BLK), 128, 0, stream>>>(
        Xb, W1t, MROWS, HID, CIN, b1, nullptr, nullptr, Hb, nullptr);

    // 5. GEMM2 + bias + residual -> out (fp32)
    gemm_bf16_wmma<1><<<dim3(CIN / BLK, MROWS / BLK), 128, 0, stream>>>(
        Hb, W2t, MROWS, CIN, HID, b2, in_tensor, gamma, nullptr, out);
}
